// MyModel_61933428415074
// MI455X (gfx1250) — 8-bit matrix operand order repaired and checked by reading, compiled, not run on hardware
//
#include <hip/hip_runtime.h>
#include <cstdint>

typedef __attribute__((ext_vector_type(16))) int   v16i;
typedef __attribute__((ext_vector_type(8)))  float v8f;

#define IN_F   784
#define K_PAD  896
#define HIDN   128
#define OUT_N  10
#define KTILES 7

__device__ __forceinline__ uint32_t pack4_sign(float f0, float f1, float f2, float f3) {
    uint32_t d = 0x38383838u;
    d |= (__float_as_uint(f0) >> 31) << 7;
    d |= (__float_as_uint(f1) >> 31) << 15;
    d |= (__float_as_uint(f2) >> 31) << 23;
    d |= (__float_as_uint(f3) & 0x80000000u);
    return d;
}

__global__ void bnn_pack_weights(const float* __restrict__ W1,
                                 const float* __restrict__ W2,
                                 uint32_t* __restrict__ w1f8,
                                 uint32_t* __restrict__ w2f8) {
    const int W1D = 128 * (K_PAD / 4);
    const int W2D = 16 * (HIDN / 4);
    int j = blockIdx.x * blockDim.x + threadIdx.x;
    if (j < W1D) {
        int row = j / (K_PAD / 4);
        int c4  = (j % (K_PAD / 4)) * 4;
        uint32_t d = 0;
        #pragma unroll
        for (int i = 0; i < 4; ++i) {
            int k = c4 + i;
            uint32_t b = 0;
            if (k < IN_F) {
                uint32_t s = __float_as_uint(W1[row * IN_F + k]) >> 31;
                b = 0x38u | (s << 7);
            }
            d |= b << (8 * i);
        }
        w1f8[j] = d;
    } else if (j < W1D + W2D) {
        int jj  = j - W1D;
        int row = jj / (HIDN / 4);
        int c4  = (jj % (HIDN / 4)) * 4;
        uint32_t d = 0;
        if (row < OUT_N) {
            #pragma unroll
            for (int i = 0; i < 4; ++i) {
                uint32_t s = __float_as_uint(W2[row * HIDN + c4 + i]) >> 31;
                d |= (0x38u | (s << 7)) << (8 * i);
            }
        }
        w2f8[jj] = d;
    }
}

__global__ __launch_bounds__(256, 2)
void bnn_fused(const float* __restrict__ x,
               const float* __restrict__ b1,
               const float* __restrict__ b2,
               const uint32_t* __restrict__ w1f8,
               const uint32_t* __restrict__ w2f8,
               float* __restrict__ out) {
    __shared__ __align__(16) unsigned char shH[8][16][HIDN];

    const int tid    = threadIdx.x;
    const int wave   = tid >> 5;
    const int lane   = tid & 31;
    const int laneLo = lane & 15;
    const int hiHalf = (lane >> 4) & 1;
    const int mbase  = blockIdx.x * 128 + wave * 16;

    const float* xrow = x + (size_t)(mbase + laneLo) * IN_F;

    v8f acc[8];
    #pragma unroll
    for (int nt = 0; nt < 8; ++nt)
        #pragma unroll
        for (int e = 0; e < 8; ++e) acc[nt][e] = 0.0f;

    for (int kt = 0; kt < KTILES; ++kt) {
        v16i a;
        const int kco = kt * 128 + hiHalf * 8;
        #pragma unroll
        for (int p = 0; p < 8; ++p) {
            int kb = kco + 16 * p;
            if (kb + 8 <= IN_F) {
                const float4 f0 = *(const float4*)(xrow + kb);
                const float4 f1 = *(const float4*)(xrow + kb + 4);
                a[2 * p]     = (int)pack4_sign(f0.x, f0.y, f0.z, f0.w);
                a[2 * p + 1] = (int)pack4_sign(f1.x, f1.y, f1.z, f1.w);
            } else {
                a[2 * p] = 0; a[2 * p + 1] = 0;
            }
        }
        #pragma unroll
        for (int nt = 0; nt < 8; ++nt) {
            const uint32_t* w1row = w1f8 + (size_t)(nt * 16 + laneLo) * (K_PAD / 4);
            v16i b;
            #pragma unroll
            for (int g = 0; g < 8; ++g) {
                int kb = kt * 128 + g * 16 + hiHalf * 8;
                const uint2 q = *(const uint2*)(w1row + (kb >> 2));
                b[2 * g + 0] = (int)q.x;  b[2 * g + 1] = (int)q.y;
            }
            acc[nt] = __builtin_amdgcn_wmma_f32_16x16x128_fp8_fp8(
                a, b, (short)0, acc[nt], false, false);
        }
    }

    #pragma unroll
    for (int nt = 0; nt < 8; ++nt) {
        int n = nt * 16 + laneLo;
        float bias = b1[n];
        #pragma unroll
        for (int r = 0; r < 8; ++r) {
            float h = acc[nt][r] + bias;
            shH[wave][hiHalf * 8 + r][n] =
                (unsigned char)(0x38u | ((__float_as_uint(h) >> 31) << 7));
        }
    }
    __syncthreads();

    v16i a2;
    #pragma unroll
    for (int p = 0; p < 8; ++p) {
        const uint2 q = *(const uint2*)&shH[wave][laneLo][16 * p + hiHalf * 8];
        a2[2 * p] = (int)q.x;  a2[2 * p + 1] = (int)q.y;
    }
    v16i bw2;
    {
        const uint32_t* w2row = w2f8 + (size_t)laneLo * (HIDN / 4);
        #pragma unroll
        for (int g = 0; g < 8; ++g) {
            int kb = g * 16 + hiHalf * 8;
            const uint2 q = *(const uint2*)(w2row + (kb >> 2));
            bw2[2 * g + 0] = (int)q.x;  bw2[2 * g + 1] = (int)q.y;
        }
    }
    v8f c2;
    #pragma unroll
    for (int e = 0; e < 8; ++e) c2[e] = 0.0f;
    v8f o = __builtin_amdgcn_wmma_f32_16x16x128_fp8_fp8(a2, bw2, (short)0, c2, false, false);

    int n = laneLo;
    if (n < OUT_N) {
        float bias = b2[n];
        #pragma unroll
        for (int r = 0; r < 8; ++r) {
            int m = mbase + hiHalf * 8 + r;
            out[(size_t)m * OUT_N + n] = o[r] + bias;
        }
    }
}

extern "C" void kernel_launch(void* const* d_in, const int* in_sizes, int n_in,
                              void* d_out, int out_size, void* d_ws, size_t ws_size,
                              hipStream_t stream) {
    const float* x  = (const float*)d_in[0];
    const float* W1 = (const float*)d_in[1];
    const float* b1 = (const float*)d_in[2];
    const float* W2 = (const float*)d_in[3];
    const float* b2 = (const float*)d_in[4];
    float* out = (float*)d_out;

    uint32_t* w1f8 = (uint32_t*)d_ws;
    uint32_t* w2f8 = w1f8 + 128 * (K_PAD / 4);

    const int B = in_sizes[0] / IN_F;

    const int packWork = 128 * (K_PAD / 4) + 16 * (HIDN / 4);
    bnn_pack_weights<<<(packWork + 255) / 256, 256, 0, stream>>>(W1, W2, w1f8, w2f8);
    bnn_fused<<<B / 128, 256, 0, stream>>>(x, b1, b2, w1f8, w2f8, out);
}
